// VisLSTM_20418274525196
// MI455X (gfx1250) — compile-verified, matrix operand order checked by reading, not run on hardware
//
#include <hip/hip_runtime.h>

// ---------------------------------------------------------------------------
// VisLSTM for MI455X (gfx1250): bf16 WMMA GEMMs + async global->LDS staging
// + sequential LSTM scan.
// ---------------------------------------------------------------------------

typedef __attribute__((ext_vector_type(16))) __bf16 v16bf;
typedef __attribute__((ext_vector_type(8)))  float  v8f;
typedef __attribute__((ext_vector_type(4)))  int    v4i;

#define ASG __attribute__((address_space(1)))
#define ASL __attribute__((address_space(3)))

#if __has_builtin(__builtin_amdgcn_global_load_async_to_lds_b128)
#define GEMM_ASYNC_COPY 1
#else
#define GEMM_ASYNC_COPY 0
#endif

#if __has_builtin(__builtin_amdgcn_s_wait_asynccnt)
#define WAIT_ASYNC(n) __builtin_amdgcn_s_wait_asynccnt(n)
#else
#define WAIT_ASYNC(n) asm volatile("s_wait_asynccnt %0" ::"i"(n) : "memory")
#endif

#define LSTM_N    512
#define LSTM_T    48
#define LSTM_TT   49
#define LSTM_H    1024
#define LSTM_G    4096   /* 4*H */
#define LSTM_V    32000
#define LSTM_IMG  4096

// GEMM tile config: 128x128x64, 256 threads = 8 wave32.
#define BM  128
#define BN  128
#define BK  64
#define BKP 72   // +8 bf16 pad: row stride 144B -> conflict-free ds_read_b128
#define STAGE_ELEMS ((BM + BN) * BKP)          // bf16 elems per LDS stage buffer
#define GEMM_LDS_BYTES (2 * STAGE_ELEMS * 2)   // double buffered = 73728 B

// ---------------------------------------------------------------------------
// C[M,N](f32 or bf16) = A[M,K](bf16) @ W[N,K]^T(bf16)  (+ bias[N]) (+ Cadd[M,N] bf16)
// W row n == B column n, so A and B tiles share one LDS layout and loader.
// Global->LDS staging via GLOBAL_LOAD_ASYNC_TO_LDS_B128 (ASYNCcnt), double
// buffered: stage s+1 streams into LDS while stage s feeds the WMMAs.
// ---------------------------------------------------------------------------
template <bool OUT_BF16, bool HAS_ADD>
__global__ __launch_bounds__(256, 1) void gemm_bf16_nt(
    const __bf16* __restrict__ A, const __bf16* __restrict__ W,
    const float* __restrict__ bias, const __bf16* __restrict__ Cadd,
    float* __restrict__ Cf, __bf16* __restrict__ Cb, int M, int N, int K) {
  extern __shared__ __bf16 smem[];  // [2][STAGE_ELEMS]

  const int tid  = threadIdx.x;
  const int lane = tid & 31;
  const int wid  = tid >> 5;
  const int wm   = wid >> 2;   // 0..1  -> 64 rows
  const int wn   = wid & 3;    // 0..3  -> 32 cols
  const int bm0  = blockIdx.y * BM;
  const int bn0  = blockIdx.x * BN;

  // staging decomposition: 256 thr * (4+4) b128 = 128x64 bf16 per matrix
  const int c8 = tid & 7;   // 8-element K chunk
  const int r0 = tid >> 3;  // 0..31, step 32

  auto stage = [&](int k0, int b) {
    __bf16* Ab = smem + b * STAGE_ELEMS;
    __bf16* Bb = Ab + BM * BKP;
#pragma unroll
    for (int p = 0; p < 4; ++p) {
      const int row = r0 + p * 32;
      const __bf16* ga = A + (size_t)(bm0 + row) * K + k0 + c8 * 8;
      const __bf16* gb = W + (size_t)(bn0 + row) * K + k0 + c8 * 8;
      __bf16* la = Ab + row * BKP + c8 * 8;
      __bf16* lb = Bb + row * BKP + c8 * 8;
#if GEMM_ASYNC_COPY
      __builtin_amdgcn_global_load_async_to_lds_b128((ASG v4i*)ga, (ASL v4i*)la, 0, 0);
      __builtin_amdgcn_global_load_async_to_lds_b128((ASG v4i*)gb, (ASL v4i*)lb, 0, 0);
#else
      *(uint4*)la = *(const uint4*)ga;
      *(uint4*)lb = *(const uint4*)gb;
#endif
    }
  };

  const v8f vzero = {0.f, 0.f, 0.f, 0.f, 0.f, 0.f, 0.f, 0.f};
  v8f acc[4][2];
#pragma unroll
  for (int mi = 0; mi < 4; ++mi)
#pragma unroll
    for (int ni = 0; ni < 2; ++ni) acc[mi][ni] = vzero;

  // WMMA 16-bit A/B fragment addressing (ISA 7.12.2):
  // lane<16: row = lane,     K = 0..7  then 16..23
  // lane>=16: row = lane-16, K = 8..15 then 24..31
  const int lrow  = lane & 15;
  const int lkoff = (lane >> 4) << 3;

  stage(0, 0);
  int cur = 0;
  const int nStages = K / BK;
  for (int s = 0; s < nStages; ++s) {
    if (s + 1 < nStages) {
      stage((s + 1) * BK, cur ^ 1);  // 8 new async ops land in the other buffer
      WAIT_ASYNC(8);                 // oldest 8 (= stage s) have retired, in order
    } else {
      WAIT_ASYNC(0);
    }
    __syncthreads();                 // buf[cur] visible to all waves

    const __bf16* Ab = smem + cur * STAGE_ELEMS;
    const __bf16* Bb = Ab + BM * BKP;
#pragma unroll
    for (int kk = 0; kk < BK; kk += 32) {
      v16bf af[4], bfr[2];
#pragma unroll
      for (int mi = 0; mi < 4; ++mi) {
        const __bf16* p = Ab + (wm * 64 + mi * 16 + lrow) * BKP + kk + lkoff;
        ((uint4*)&af[mi])[0] = *(const uint4*)p;
        ((uint4*)&af[mi])[1] = *(const uint4*)(p + 16);
      }
#pragma unroll
      for (int ni = 0; ni < 2; ++ni) {
        const __bf16* p = Bb + (wn * 32 + ni * 16 + lrow) * BKP + kk + lkoff;
        ((uint4*)&bfr[ni])[0] = *(const uint4*)p;
        ((uint4*)&bfr[ni])[1] = *(const uint4*)(p + 16);
      }
#pragma unroll
      for (int mi = 0; mi < 4; ++mi)
#pragma unroll
        for (int ni = 0; ni < 2; ++ni)
          acc[mi][ni] = __builtin_amdgcn_wmma_f32_16x16x32_bf16(
              false, af[mi], false, bfr[ni], (short)0, acc[mi][ni], false, false);
    }
    __syncthreads();  // all waves done reading buf[cur] before it is restaged
    cur ^= 1;
  }

  // Epilogue. C/D layout: lane<16 -> N=lane, M = v; lane>=16 -> M = v+8.
  const int lhalf = lane >> 4;
#pragma unroll
  for (int mi = 0; mi < 4; ++mi) {
#pragma unroll
    for (int ni = 0; ni < 2; ++ni) {
      const int col  = bn0 + wn * 32 + ni * 16 + lrow;
      const int row0 = bm0 + wm * 64 + mi * 16 + lhalf * 8;
      const float bv = bias ? bias[col] : 0.0f;
#pragma unroll
      for (int v = 0; v < 8; ++v) {
        const size_t idx = (size_t)(row0 + v) * N + col;
        float r = acc[mi][ni][v] + bv;
        if (HAS_ADD) r += (float)Cadd[idx];
        if (OUT_BF16) Cb[idx] = (__bf16)r;
        else          Cf[idx] = r;
      }
    }
  }
}

// ---------------------------------------------------------------------------
// Elementwise / data-movement helpers
// ---------------------------------------------------------------------------
__global__ void cvt_bf16x4(const float* __restrict__ src, __bf16* __restrict__ dst, int n4) {
  const int i = blockIdx.x * blockDim.x + threadIdx.x;
  if (i >= n4) return;
  const float4 v = ((const float4*)src)[i];
  __bf16* d = dst + 4 * (size_t)i;
  d[0] = (__bf16)v.x; d[1] = (__bf16)v.y; d[2] = (__bf16)v.z; d[3] = (__bf16)v.w;
}

__global__ void bias_fold(const float* a, const float* b, float* o) {
  const int i = blockIdx.x * blockDim.x + threadIdx.x;
  o[i] = a[i] + b[i];
}

// One block per (t,n) row of the token sequence; slot depends on first_words.
__global__ void gather_emb_kernel(const int* __restrict__ q, const float* __restrict__ emb,
                                  __bf16* __restrict__ X, const int* __restrict__ fw) {
  const int rid = blockIdx.x;          // 0 .. T*N-1
  const int t   = rid >> 9;            // /512
  const int n   = rid & 511;
  const int slot = (*fw != 0) ? (t + 1) : t;
  const int tok  = q[n * LSTM_T + t];
  const float4* s = (const float4*)(emb + (size_t)tok * LSTM_H);
  __bf16* d = X + ((size_t)slot * LSTM_N + n) * LSTM_H;
  const float4 v = s[threadIdx.x];     // 256 thr * 4 = 1024 elems
  const int e = threadIdx.x * 4;
  d[e + 0] = (__bf16)v.x; d[e + 1] = (__bf16)v.y;
  d[e + 2] = (__bf16)v.z; d[e + 3] = (__bf16)v.w;
}

__global__ void place_img_kernel(const __bf16* __restrict__ tmp, __bf16* __restrict__ X,
                                 const int* __restrict__ fw) {
  const int slot = (*fw != 0) ? 0 : LSTM_T;
  const size_t i = (size_t)blockIdx.x * blockDim.x + threadIdx.x;
  X[(size_t)slot * LSTM_N * LSTM_H + i] = tmp[i];
}

__global__ void init_state(float* __restrict__ c, __bf16* __restrict__ h) {
  const size_t i = (size_t)blockIdx.x * blockDim.x + threadIdx.x;
  c[i] = 0.0f;
  h[i] = (__bf16)0.0f;
}

__device__ __forceinline__ float fast_sigmoid(float x) {
  return 1.0f / (1.0f + __expf(-x));
}
__device__ __forceinline__ float fast_tanh(float x) {
  const float e = __expf(-2.0f * x);
  return (1.0f - e) / (1.0f + e);
}

// gates[N,4H] (i,f,g,o torch order) -> update c, write h (fp32 to hidden_states,
// bf16 for next-step GEMM).
__global__ void lstm_cell_kernel(const float* __restrict__ gates, float* __restrict__ c,
                                 __bf16* __restrict__ h_bf, float* __restrict__ hs, int t) {
  const size_t idx = (size_t)blockIdx.x * blockDim.x + threadIdx.x;  // 0..N*H-1
  const int n = (int)(idx >> 10);
  const int j = (int)(idx & 1023);
  const float* g = gates + (size_t)n * LSTM_G;
  const float gi = fast_sigmoid(g[j]);
  const float gf = fast_sigmoid(g[LSTM_H + j]);
  const float gg = fast_tanh(g[2 * LSTM_H + j]);
  const float go = fast_sigmoid(g[3 * LSTM_H + j]);
  const float cc = gf * c[idx] + gi * gg;
  c[idx] = cc;
  const float h = go * fast_tanh(cc);
  h_bf[idx] = (__bf16)h;
  hs[((size_t)n * LSTM_TT + t) * LSTM_H + j] = h;
}

// ---------------------------------------------------------------------------
// Launcher
// ---------------------------------------------------------------------------
extern "C" void kernel_launch(void* const* d_in, const int* in_sizes, int n_in,
                              void* d_out, int out_size, void* d_ws, size_t ws_size,
                              hipStream_t stream) {
  const int*   questions = (const int*)  d_in[0];
  const float* img       = (const float*)d_in[1];
  const float* emb       = (const float*)d_in[2];
  const float* W_img     = (const float*)d_in[3];
  const float* b_img     = (const float*)d_in[4];
  const float* W_ih      = (const float*)d_in[5];
  const float* W_hh      = (const float*)d_in[6];
  const float* b_ih      = (const float*)d_in[7];
  const float* b_hh      = (const float*)d_in[8];
  const float* W_out     = (const float*)d_in[9];
  const float* b_out     = (const float*)d_in[10];
  const int*   fw        = (const int*)  d_in[11];

  float* out_logits = (float*)d_out;                                  // [512,32000]
  float* out_hidden = (float*)d_out + (size_t)LSTM_N * LSTM_V;        // [512,49,1024]

  char* ws = (char*)d_ws;
  size_t off = 0;
  auto alloc = [&](size_t bytes) -> void* {
    void* p = ws + off;
    off = (off + bytes + 255) & ~(size_t)255;
    return p;
  };
  __bf16* Wih_b  = (__bf16*)alloc((size_t)LSTM_G * LSTM_H * 2);
  __bf16* Whh_b  = (__bf16*)alloc((size_t)LSTM_G * LSTM_H * 2);
  __bf16* Wimg_b = (__bf16*)alloc((size_t)LSTM_H * LSTM_IMG * 2);
  __bf16* Wout_b = (__bf16*)alloc((size_t)LSTM_V * LSTM_H * 2);
  __bf16* img_b  = (__bf16*)alloc((size_t)LSTM_N * LSTM_IMG * 2);
  __bf16* Xb     = (__bf16*)alloc((size_t)LSTM_TT * LSTM_N * LSTM_H * 2);
  __bf16* XW     = (__bf16*)alloc((size_t)LSTM_TT * LSTM_N * LSTM_G * 2);
  float*  gates  = (float*) alloc((size_t)LSTM_N * LSTM_G * 4);
  float*  c_st   = (float*) alloc((size_t)LSTM_N * LSTM_H * 4);
  __bf16* h_b    = (__bf16*)alloc((size_t)LSTM_N * LSTM_H * 2);
  __bf16* imgemb = (__bf16*)alloc((size_t)LSTM_N * LSTM_H * 2);
  float*  bias_c = (float*) alloc((size_t)LSTM_G * 4);

  const int thr = 256;
  auto cvt = [&](const float* s, __bf16* d, size_t n) {
    const int n4 = (int)(n / 4);
    cvt_bf16x4<<<(n4 + thr - 1) / thr, thr, 0, stream>>>(s, d, n4);
  };
  // 1) weight / activation down-conversion (cheap, fully parallel)
  cvt(W_ih,  Wih_b,  (size_t)LSTM_G * LSTM_H);
  cvt(W_hh,  Whh_b,  (size_t)LSTM_G * LSTM_H);
  cvt(W_img, Wimg_b, (size_t)LSTM_H * LSTM_IMG);
  cvt(W_out, Wout_b, (size_t)LSTM_V * LSTM_H);
  cvt(img,   img_b,  (size_t)LSTM_N * LSTM_IMG);
  bias_fold<<<LSTM_G / thr, thr, 0, stream>>>(b_ih, b_hh, bias_c);

  // 2) sequence assembly (token gather + image projection GEMM)
  gather_emb_kernel<<<LSTM_T * LSTM_N, thr, 0, stream>>>(questions, emb, Xb, fw);
  gemm_bf16_nt<true, false>
      <<<dim3(LSTM_H / BN, LSTM_N / BM), thr, GEMM_LDS_BYTES, stream>>>(
          img_b, Wimg_b, b_img, nullptr, nullptr, imgemb, LSTM_N, LSTM_H, LSTM_IMG);
  place_img_kernel<<<(LSTM_N * LSTM_H) / thr, thr, 0, stream>>>(imgemb, Xb, fw);

  // 3) hoisted input-gate GEMM: XW[t] = x_t @ W_ih^T + (b_ih + b_hh)
  gemm_bf16_nt<true, false>
      <<<dim3(LSTM_G / BN, (LSTM_TT * LSTM_N) / BM), thr, GEMM_LDS_BYTES, stream>>>(
          Xb, Wih_b, bias_c, nullptr, nullptr, XW, LSTM_TT * LSTM_N, LSTM_G, LSTM_H);

  // 4) recurrent scan: only h @ W_hh^T on the critical path
  init_state<<<(LSTM_N * LSTM_H) / thr, thr, 0, stream>>>(c_st, h_b);
  for (int t = 0; t < LSTM_TT; ++t) {
    gemm_bf16_nt<false, true>
        <<<dim3(LSTM_G / BN, LSTM_N / BM), thr, GEMM_LDS_BYTES, stream>>>(
            h_b, Whh_b, nullptr, XW + (size_t)t * LSTM_N * LSTM_G, gates, nullptr,
            LSTM_N, LSTM_G, LSTM_H);
    lstm_cell_kernel<<<(LSTM_N * LSTM_H) / thr, thr, 0, stream>>>(
        gates, c_st, h_b, out_hidden, t);
  }

  // 5) output head straight into d_out
  gemm_bf16_nt<false, false>
      <<<dim3(LSTM_V / BN, LSTM_N / BM), thr, GEMM_LDS_BYTES, stream>>>(
          h_b, Wout_b, b_out, nullptr, out_logits, nullptr, LSTM_N, LSTM_V, LSTM_H);
}
